// SageConv_81527069213077
// MI455X (gfx1250) — matrix operand order repaired and checked by reading, not compiled, not run on hardware
//
#include <hip/hip_runtime.h>

// SageConv on MI455X (gfx1250): adj@features is HBM-bound on the 1 GiB adj
// read; use bf16 WMMA (fp32 accum) so compute is never the bottleneck.
// features/W are pre-packed into WMMA B-fragment layout in d_ws (L2-resident).

typedef __attribute__((ext_vector_type(16))) __bf16 v16bf;
typedef __attribute__((ext_vector_type(8)))  float  v8f;
typedef __attribute__((ext_vector_type(4)))  float  f4;

#define N_NODES 16384
#define KB_TOTAL (N_NODES / 32)   // 512 K-blocks of 32

__device__ __forceinline__ unsigned pk_bf16(float lo, float hi) {
  unsigned short l = __builtin_bit_cast(unsigned short, (__bf16)lo);
  unsigned short h = __builtin_bit_cast(unsigned short, (__bf16)hi);
  return (unsigned)l | ((unsigned)h << 16);
}

// Build a 16x32 bf16 A-fragment (contiguous K order, K rel = half*16 + s) from two 8-float runs:
// f* = K rel {half*16 .. half*16+7}, g* = K rel {half*16+8 .. half*16+15}.
__device__ __forceinline__ v16bf cvt_frag(f4 f0, f4 f1, f4 g0, f4 g1) {
  v16bf a;
  a[0]  = (__bf16)f0.x; a[1]  = (__bf16)f0.y; a[2]  = (__bf16)f0.z; a[3]  = (__bf16)f0.w;
  a[4]  = (__bf16)f1.x; a[5]  = (__bf16)f1.y; a[6]  = (__bf16)f1.z; a[7]  = (__bf16)f1.w;
  a[8]  = (__bf16)g0.x; a[9]  = (__bf16)g0.y; a[10] = (__bf16)g0.z; a[11] = (__bf16)g0.w;
  a[12] = (__bf16)g1.x; a[13] = (__bf16)g1.y; a[14] = (__bf16)g1.z; a[15] = (__bf16)g1.w;
  return a;
}

// ---- Prep: pack features [16384,64] fp32 -> bf16 B-fragments in ws.
// Fragment (kb, nt): 32x16 bf16 B tile; per lane 8 dwords stored contiguously
// at dword offset ((kb*4+nt)*32 + lane)*8.  B layout: lane half -> K +16,
// dword v -> K = 2v,2v+1 ; N = nt*16 + (lane&15).
__global__ void __launch_bounds__(256) pack_b_features(
    const float* __restrict__ F, unsigned* __restrict__ out) {
  int t = blockIdx.x * 256 + threadIdx.x;          // [0, 512*4*32)
  if (t >= KB_TOTAL * 4 * 32) return;
  int lane = t & 31;
  int nt   = (t >> 5) & 3;
  int kb   = t >> 7;
  int n     = nt * 16 + (lane & 15);
  int kbase = kb * 32 + ((lane >> 4) * 16);
  unsigned* dst = out + (size_t)t * 8;
#pragma unroll
  for (int v = 0; v < 8; ++v) {
    float lo = F[(size_t)(kbase + 2 * v)     * 64 + n];
    float hi = F[(size_t)(kbase + 2 * v + 1) * 64 + n];
    dst[v] = pk_bf16(lo, hi);
  }
}

// ---- Prep: pack W [64,128] as B[k,n] = W[n,k] -> 16 fragments (4 kb x 4 nt).
__global__ void __launch_bounds__(256) pack_b_weight(
    const float* __restrict__ W, unsigned* __restrict__ out) {
  int t = blockIdx.x * 256 + threadIdx.x;          // [0, 4*4*32)
  if (t >= 4 * 4 * 32) return;
  int lane = t & 31;
  int nt   = (t >> 5) & 3;
  int kb   = t >> 7;
  int n     = nt * 16 + (lane & 15);
  int kbase = kb * 32 + ((lane >> 4) * 16);
  unsigned* dst = out + (size_t)t * 8;
#pragma unroll
  for (int v = 0; v < 8; ++v) {
    float lo = W[(size_t)n * 128 + kbase + 2 * v];
    float hi = W[(size_t)n * 128 + kbase + 2 * v + 1];
    dst[v] = pk_bf16(lo, hi);
  }
}

// ---- Main pass: neigh[m0:m0+16, 0:64] = (adj @ features) / (rowsum(adj)+1).
// 1024 blocks x 8 waves; each wave owns K-blocks kb = wave, wave+8, ...
// adj streamed fp32 -> bf16 in-register; 4 WMMA (one per 16-col tile) per step.
__global__ void __launch_bounds__(256) sage_aggregate(
    const float* __restrict__ adj, const unsigned* __restrict__ bfeat,
    float* __restrict__ neigh) {
  __shared__ float s_part[8 * 16 * 64];   // per-wave partial C tiles (32 KB)
  __shared__ float s_degp[8 * 32];        // per-(wave,lane) degree partials
  __shared__ float s_deg[16];

  int tid  = threadIdx.x;
  int wave = tid >> 5;
  int lane = tid & 31;
  int half = lane >> 4;
  int m0   = blockIdx.x * 16;
  int row  = m0 + (lane & 15);

  const float* arow = adj + (size_t)row * N_NODES + half * 16;
  const v16bf* bv   = (const v16bf*)bfeat;

  v8f acc0 = {}, acc1 = {}, acc2 = {}, acc3 = {};
  float deg = 0.0f;

  for (int kb = wave; kb < KB_TOTAL; kb += 8) {
    const float* p = arow + kb * 32;
    __builtin_prefetch(p + 8 * 32, 0, 1);          // next tile for this wave
    f4 f0 = *(const f4*)(p);
    f4 f1 = *(const f4*)(p + 4);
    f4 g0 = *(const f4*)(p + 8);
    f4 g1 = *(const f4*)(p + 12);
    deg += f0.x + f0.y + f0.z + f0.w + f1.x + f1.y + f1.z + f1.w
         + g0.x + g0.y + g0.z + g0.w + g1.x + g1.y + g1.z + g1.w;
    v16bf a = cvt_frag(f0, f1, g0, g1);
    size_t bb = (size_t)(kb * 4) * 32 + lane;
    acc0 = __builtin_amdgcn_wmma_f32_16x16x32_bf16(false, a, false, bv[bb +  0],
                                                   (short)0, acc0, false, false);
    acc1 = __builtin_amdgcn_wmma_f32_16x16x32_bf16(false, a, false, bv[bb + 32],
                                                   (short)0, acc1, false, false);
    acc2 = __builtin_amdgcn_wmma_f32_16x16x32_bf16(false, a, false, bv[bb + 64],
                                                   (short)0, acc2, false, false);
    acc3 = __builtin_amdgcn_wmma_f32_16x16x32_bf16(false, a, false, bv[bb + 96],
                                                   (short)0, acc3, false, false);
  }

  // Deterministic cross-wave reduction through LDS (fixed order).
  {
    int base = wave * 1024;
    int n0   = lane & 15;
#pragma unroll
    for (int v = 0; v < 8; ++v) {
      int m = v + half * 8;
      s_part[base + m * 64 +  0 + n0] = acc0[v];
      s_part[base + m * 64 + 16 + n0] = acc1[v];
      s_part[base + m * 64 + 32 + n0] = acc2[v];
      s_part[base + m * 64 + 48 + n0] = acc3[v];
    }
    s_degp[wave * 32 + lane] = deg;
  }
  __syncthreads();
  if (tid < 16) {
    float d = 1.0f;
#pragma unroll
    for (int w = 0; w < 8; ++w)
      d += s_degp[w * 32 + tid] + s_degp[w * 32 + tid + 16];
    s_deg[tid] = d;
  }
  __syncthreads();
  for (int idx = tid; idx < 16 * 64; idx += 256) {
    float s = 0.0f;
#pragma unroll
    for (int w = 0; w < 8; ++w) s += s_part[w * 1024 + idx];
    int m = idx >> 6;
    neigh[(size_t)(m0 + m) * 64 + (idx & 63)] = s / s_deg[m];
  }
}

// ---- Output GEMM: out = [features | neigh] @ W^T.  1024 blocks x 4 waves;
// wave nt owns output columns [nt*16, nt*16+16); K = 128 in 4 WMMA steps.
__global__ void __launch_bounds__(128) sage_output(
    const float* __restrict__ feat, const float* __restrict__ neigh,
    const unsigned* __restrict__ bw, float* __restrict__ out) {
  int tid  = threadIdx.x;
  int nt   = tid >> 5;
  int lane = tid & 31;
  int half = lane >> 4;
  int m0   = blockIdx.x * 16;
  int row  = m0 + (lane & 15);

  const v16bf* bv = (const v16bf*)bw;
  v8f acc = {};
#pragma unroll
  for (int kb = 0; kb < 4; ++kb) {
    const float* src = (kb < 2) ? feat : neigh;       // data = [feat | neigh]
    int colb = kb * 32 - ((kb < 2) ? 0 : 64) + half * 16;
    const float* p = src + (size_t)row * 64 + colb;
    f4 f0 = *(const f4*)(p);
    f4 f1 = *(const f4*)(p + 4);
    f4 g0 = *(const f4*)(p + 8);
    f4 g1 = *(const f4*)(p + 12);
    v16bf a = cvt_frag(f0, f1, g0, g1);
    acc = __builtin_amdgcn_wmma_f32_16x16x32_bf16(
        false, a, false, bv[(size_t)(kb * 4 + nt) * 32 + lane],
        (short)0, acc, false, false);
  }
#pragma unroll
  for (int v = 0; v < 8; ++v)
    out[(size_t)(m0 + v + half * 8) * 64 + nt * 16 + (lane & 15)] = acc[v];
}

extern "C" void kernel_launch(void* const* d_in, const int* in_sizes, int n_in,
                              void* d_out, int out_size, void* d_ws, size_t ws_size,
                              hipStream_t stream) {
  (void)in_sizes; (void)n_in; (void)out_size; (void)ws_size;
  const float* features = (const float*)d_in[0];   // [16384, 64]
  const float* adj      = (const float*)d_in[1];   // [16384, 16384]
  const float* W        = (const float*)d_in[2];   // [64, 128]
  float* out = (float*)d_out;                      // [16384, 64]

  // Workspace layout (~6.02 MB): bf16 feature B-frags | bf16 W B-frags | neigh
  unsigned* bfeat = (unsigned*)d_ws;               // 512*4*32*8 dw = 2 MB
  unsigned* bw    = bfeat + (size_t)KB_TOTAL * 4 * 32 * 8;  // 4*4*32*8 dw = 16 KB
  float*    neigh = (float*)(bw + 4 * 4 * 32 * 8); // 16384*64 fp32 = 4 MB

  pack_b_features<<<(KB_TOTAL * 4 * 32 + 255) / 256, 256, 0, stream>>>(features, bfeat);
  pack_b_weight<<<2, 256, 0, stream>>>(W, bw);
  sage_aggregate<<<N_NODES / 16, 256, 0, stream>>>(adj, bfeat, neigh);
  sage_output<<<N_NODES / 16, 128, 0, stream>>>(features, neigh, bw, out);
}
